// EdgePredictor_34522947125625
// MI455X (gfx1250) — matrix operand order repaired and checked by reading, not compiled, not run on hardware
//
#include <hip/hip_runtime.h>
#include <math.h>

// ---------- problem constants ----------
#define BB    8
#define NN    192
#define RR    (BB*NN)      // 1536 rows
#define HH    512
#define H2    256
#define H4    128
#define PPAIR 18336        // C(192,2)
#define NEDGE (BB*PPAIR)   // 146688

// ---------- WMMA helper types ----------
typedef __attribute__((ext_vector_type(16))) _Float16 v16h;
typedef __attribute__((ext_vector_type(8)))  float    v8f;
typedef __attribute__((ext_vector_type(4)))  unsigned uint4v;

__device__ inline v8f wmma_f16(v16h a, v16h b, v8f c) {
  // D = A(16x32 f16) x B(32x16 f16) + C(16x16 f32)
  return __builtin_amdgcn_wmma_f32_16x16x32_f16(
      false, a, false, b, (short)0, c, false, false);
}

// A fragment: 16x32 tile, rows row-major with 'stride' halfs.
// ISA layout: lanes 0-15 row M=lane, K = k0+{0..7, 16..23}; lanes 16-31 K += 8.
__device__ inline v16h frag_A(const _Float16* base, int stride, int k0) {
  int lane = threadIdx.x & 31;
  const _Float16* p = base + (size_t)(lane & 15) * stride + k0 + (lane >> 4) * 8;
  v16h r;
  ((uint4v*)&r)[0] = *(const uint4v*)p;          // K offs 0..7
  ((uint4v*)&r)[1] = *(const uint4v*)(p + 16);   // K offs 16..23
  return r;
}

// B fragment: 32x16 tile from TRANSPOSED weight Wt[N][K] (K contiguous per n).
// ISA layout: lanes 0-15 col N=n0+lane, K = k0+{0..7, 16..23}; lanes 16-31 K += 8.
__device__ inline v16h frag_B(const _Float16* Wt, int Kstride, int n0, int k0) {
  int lane = threadIdx.x & 31;
  const _Float16* p = Wt + (size_t)(n0 + (lane & 15)) * Kstride + k0 + (lane >> 4) * 8;
  v16h r;
  ((uint4v*)&r)[0] = *(const uint4v*)p;          // K offs 0..7
  ((uint4v*)&r)[1] = *(const uint4v*)(p + 16);   // K offs 16..23
  return r;
}

__device__ inline float gelu_f(float x) {
  return 0.5f * x * (1.f + erff(x * 0.70710678118654752f));
}

// ---------- tiny prep kernels ----------
__global__ __launch_bounds__(256) void k_pairs(int* iu, int* ju) {
  int p = blockIdx.x * 256 + threadIdx.x;
  if (p >= PPAIR) return;
  int i = 0, rem = p;
  while (rem >= (NN - 1 - i)) { rem -= (NN - 1 - i); ++i; }
  iu[p] = i; ju[p] = i + 1 + rem;
}

// dst[n*K+k] = (f16)src[k*N+n]   (transpose + convert, src K rows x N cols)
__global__ __launch_bounds__(256) void k_t16(const float* __restrict__ src,
                                             _Float16* __restrict__ dst,
                                             int K, int N) {
  int idx = blockIdx.x * 256 + threadIdx.x;
  if (idx >= K * N) return;
  int k = idx / N, n = idx - k * N;
  dst[(size_t)n * K + k] = (_Float16)src[idx];
}

// x = gelu(LN(vertices@W1 + b1)) -> f16, one block per row, 256 cols
__global__ __launch_bounds__(256) void k_vertex1(const float* __restrict__ vert,
                                                 const float* __restrict__ W1,
                                                 const float* __restrict__ b1,
                                                 const float* __restrict__ g,
                                                 const float* __restrict__ bb,
                                                 _Float16* __restrict__ xh) {
  int r = blockIdx.x, t = threadIdx.x;
  float v0 = vert[r*3+0], v1 = vert[r*3+1], v2 = vert[r*3+2];
  float y = v0*W1[t] + v1*W1[256+t] + v2*W1[512+t] + b1[t];
  __shared__ float rs[256], rs2[256];
  rs[t] = y; rs2[t] = y*y; __syncthreads();
  for (int o = 128; o > 0; o >>= 1) {
    if (t < o) { rs[t] += rs[t+o]; rs2[t] += rs2[t+o]; }
    __syncthreads();
  }
  float mean = rs[0] * (1.f/256.f);
  float var  = rs2[0] * (1.f/256.f) - mean*mean;
  float val  = (y - mean) * rsqrtf(var + 1e-5f) * g[t] + bb[t];
  xh[(size_t)r*256 + t] = (_Float16)gelu_f(val);
}

// row-wise LayerNorm: f32 in -> f32 out (+optional f16 copy)
__global__ __launch_bounds__(256) void k_ln(const float* __restrict__ Y,
                                            const float* __restrict__ g,
                                            const float* __restrict__ b,
                                            float* __restrict__ Of,
                                            _Float16* __restrict__ Oh, int W) {
  int r = blockIdx.x, t = threadIdx.x;
  float s = 0.f, s2 = 0.f;
  for (int c = t; c < W; c += 256) { float v = Y[(size_t)r*W+c]; s += v; s2 += v*v; }
  __shared__ float rs[256], rs2[256];
  rs[t] = s; rs2[t] = s2; __syncthreads();
  for (int o = 128; o > 0; o >>= 1) {
    if (t < o) { rs[t] += rs[t+o]; rs2[t] += rs2[t+o]; }
    __syncthreads();
  }
  float mean = rs[0] / (float)W;
  float var  = rs2[0] / (float)W - mean*mean;
  float inv  = rsqrtf(var + 1e-5f);
  for (int c = t; c < W; c += 256) {
    float v = (Y[(size_t)r*W+c] - mean) * inv * g[c] + b[c];
    Of[(size_t)r*W+c] = v;
    if (Oh) Oh[(size_t)r*W+c] = (_Float16)v;
  }
}

// Cpre/Dpre: vertices @ We1 rows 1024..1026 / 1027..1029
__global__ __launch_bounds__(256) void k_cdpre(const float* __restrict__ vert,
                                               const float* __restrict__ We1,
                                               float* __restrict__ Cp,
                                               float* __restrict__ Dp) {
  int r = blockIdx.x, t = threadIdx.x;
  float v0 = vert[r*3+0], v1 = vert[r*3+1], v2 = vert[r*3+2];
  for (int c = t; c < HH; c += 256) {
    Cp[(size_t)r*HH+c] = v0*We1[(size_t)1024*HH+c] + v1*We1[(size_t)1025*HH+c] + v2*We1[(size_t)1026*HH+c];
    Dp[(size_t)r*HH+c] = v0*We1[(size_t)1027*HH+c] + v1*We1[(size_t)1028*HH+c] + v2*We1[(size_t)1029*HH+c];
  }
}

// ---------- generic WMMA GEMM: Yf = Xh(MxK f16) @ Wt(NxK f16,T) + bias (+resid) ----------
__global__ __launch_bounds__(128) void k_gemm(const _Float16* __restrict__ Xh,
                                              const _Float16* __restrict__ Wt,
                                              const float* __restrict__ bias,
                                              const float* __restrict__ resid,
                                              float* __restrict__ Yf,
                                              _Float16* __restrict__ Yh,
                                              int K, int N) {
  int m0 = blockIdx.x * 16;
  int n0 = blockIdx.y * 64 + (threadIdx.x >> 5) * 16;
  const _Float16* arow = Xh + (size_t)m0 * K;
  v8f acc = {};
  for (int k0 = 0; k0 < K; k0 += 32) {
    v16h a = frag_A(arow, K, k0);
    v16h b = frag_B(Wt, K, n0, k0);
    acc = wmma_f16(a, b, acc);
  }
  int lane = threadIdx.x & 31;
  int nc = n0 + (lane & 15);
  int mb = m0 + ((lane < 16) ? 0 : 8);
  float bv = bias ? bias[nc] : 0.f;
#pragma unroll
  for (int r = 0; r < 8; ++r) {
    size_t o = (size_t)(mb + r) * N + nc;
    float v = acc[r] + bv;
    if (resid) v += resid[o];
    Yf[o] = v;
    if (Yh) Yh[o] = (_Float16)v;
  }
}

// ---------- attention (VALU; ~0.6 GFLOPs total) ----------
__global__ __launch_bounds__(256) void k_attn(const float* __restrict__ q,
                                              const float* __restrict__ k,
                                              const float* __restrict__ v,
                                              _Float16* __restrict__ ctxh) {
  int w    = threadIdx.x >> 5;
  int lane = threadIdx.x & 31;
  int wid  = blockIdx.x * 8 + w;            // one wave per (b,h,query)
  int b  = wid / (8 * NN);
  int rm = wid - b * (8 * NN);
  int h  = rm / NN;
  int qi = rm - h * NN;
  const float* qrow = q + ((size_t)(b*NN + qi) * HH) + h * 64;
  float s[6];
#pragma unroll
  for (int t = 0; t < 6; ++t) {
    int key = lane + 32*t;
    const float* krow = k + ((size_t)(b*NN + key) * HH) + h * 64;
    float a = 0.f;
    for (int d = 0; d < 64; ++d) a += qrow[d] * krow[d];
    s[t] = a * 0.125f;                       // 1/sqrt(64)
  }
  float mx = s[0];
#pragma unroll
  for (int t = 1; t < 6; ++t) mx = fmaxf(mx, s[t]);
  for (int o = 16; o > 0; o >>= 1) mx = fmaxf(mx, __shfl_xor(mx, o, 32));
  float sum = 0.f;
#pragma unroll
  for (int t = 0; t < 6; ++t) { s[t] = expf(s[t] - mx); sum += s[t]; }
  for (int o = 16; o > 0; o >>= 1) sum += __shfl_xor(sum, o, 32);
  float inv = 1.f / sum;
  __shared__ float attn[8][NN];
#pragma unroll
  for (int t = 0; t < 6; ++t) attn[w][lane + 32*t] = s[t] * inv;
  for (int d = lane; d < 64; d += 32) {
    float a = 0.f;
    for (int key = 0; key < NN; ++key)
      a += attn[w][key] * v[((size_t)(b*NN + key) * HH) + h * 64 + d];
    ctxh[((size_t)(b*NN + qi) * HH) + h * 64 + d] = (_Float16)a;
  }
}

// ---------- fused edge kernel: 16 edges / block, 4 waves ----------
__device__ inline void ln_stats16(const float* base, int W, float* smean, float* sinv) {
  int tid = threadIdx.x;
  int e = tid >> 3, part = tid & 7;          // 8 threads per edge
  float s = 0.f, s2 = 0.f;
  for (int c = part; c < W; c += 8) { float v = base[(size_t)e*W + c]; s += v; s2 += v*v; }
  s += __shfl_xor(s, 1, 32); s2 += __shfl_xor(s2, 1, 32);
  s += __shfl_xor(s, 2, 32); s2 += __shfl_xor(s2, 2, 32);
  s += __shfl_xor(s, 4, 32); s2 += __shfl_xor(s2, 4, 32);
  if (part == 0) {
    float mean = s / (float)W;
    float var  = s2 / (float)W - mean*mean;
    smean[e] = mean; sinv[e] = rsqrtf(var + 1e-5f);
  }
  __syncthreads();
}

__global__ __launch_bounds__(128) void k_edge(
    const float* __restrict__ Apre, const float* __restrict__ Bpre,
    const float* __restrict__ Cpre, const float* __restrict__ Dpre,
    const float* __restrict__ w1030,
    const float* __restrict__ lne1g, const float* __restrict__ lne1b,
    const _Float16* __restrict__ We2t, const float* __restrict__ be2,
    const float* __restrict__ lne2g, const float* __restrict__ lne2b,
    const _Float16* __restrict__ We3t, const float* __restrict__ be3,
    const float* __restrict__ We4, const float* __restrict__ be4,
    const float* __restrict__ vert, const int* __restrict__ iu,
    const int* __restrict__ ju, float* __restrict__ out) {
  __shared__ __align__(16) float    sA[16 * HH];   // e1 (16x512), later y2 (16x256 f32)
  __shared__ __align__(16) _Float16 sB[16 * HH];   // h1 (16x512), later h3 (16x128)
  __shared__ __align__(16) _Float16 sC[16 * H2];   // h2 (16x256)
  __shared__ int   sgi[16], sgj[16];
  __shared__ float sdist[16], smean[16], sinv[16];

  int tid = threadIdx.x;
  if (tid < 16) {
    int ge = blockIdx.x * 16 + tid;
    int b = ge / PPAIR; int p = ge - b * PPAIR;
    int i = iu[p], j = ju[p];
    int gi = b * NN + i, gj = b * NN + j;
    sgi[tid] = gi; sgj[tid] = gj;
    float dx = vert[gi*3+0] - vert[gj*3+0];
    float dy = vert[gi*3+1] - vert[gj*3+1];
    float dz = vert[gi*3+2] - vert[gj*3+2];
    sdist[tid] = sqrtf(dx*dx + dy*dy + dz*dz);
  }
  __syncthreads();

  // ---- e1 = Apre[i] + Bpre[j] + Cpre[i] + Dpre[j] + dist*We1[1030]  (be1 folded in Apre)
  for (int idx = tid; idx < 16 * HH; idx += 128) {
    int e = idx >> 9, c = idx & (HH - 1);
    int gi = sgi[e], gj = sgj[e];
    sA[idx] = Apre[(size_t)gi*HH + c] + Bpre[(size_t)gj*HH + c]
            + Cpre[(size_t)gi*HH + c] + Dpre[(size_t)gj*HH + c]
            + sdist[e] * w1030[c];
  }
  __syncthreads();

  ln_stats16(sA, HH, smean, sinv);
  for (int idx = tid; idx < 16 * HH; idx += 128) {
    int e = idx >> 9, c = idx & (HH - 1);
    float v = (sA[idx] - smean[e]) * sinv[e] * lne1g[c] + lne1b[c];
    sB[idx] = (_Float16)gelu_f(v);
  }
  __syncthreads();

  // ---- GEMM1: h1(16x512) @ We2t -> y2(16x256); 4 waves x 4 col tiles
  {
    int wv = tid >> 5;
    v8f acc[4] = {{}, {}, {}, {}};
    for (int k0 = 0; k0 < HH; k0 += 32) {
      v16h a = frag_A(sB, HH, k0);
#pragma unroll
      for (int t = 0; t < 4; ++t)
        acc[t] = wmma_f16(a, frag_B(We2t, HH, (wv*4 + t) * 16, k0), acc[t]);
    }
    int lane = tid & 31;
    int nco = lane & 15;
    int mrb = (lane < 16) ? 0 : 8;
    float* y2 = sA;                          // reuse (e1 dead)
#pragma unroll
    for (int t = 0; t < 4; ++t) {
      int n = (wv*4 + t) * 16 + nco;
#pragma unroll
      for (int r = 0; r < 8; ++r)
        y2[(size_t)(r + mrb) * H2 + n] = acc[t][r] + be2[n];
    }
  }
  __syncthreads();

  ln_stats16(sA, H2, smean, sinv);
  for (int idx = tid; idx < 16 * H2; idx += 128) {
    int e = idx >> 8, c = idx & (H2 - 1);
    float v = (sA[idx] - smean[e]) * sinv[e] * lne2g[c] + lne2b[c];
    sC[idx] = (_Float16)gelu_f(v);
  }
  __syncthreads();

  // ---- GEMM2: h2(16x256) @ We3t -> h3(16x128) + gelu; 4 waves x 2 col tiles
  {
    int wv = tid >> 5;
    v8f acc[2] = {{}, {}};
    for (int k0 = 0; k0 < H2; k0 += 32) {
      v16h a = frag_A(sC, H2, k0);
#pragma unroll
      for (int t = 0; t < 2; ++t)
        acc[t] = wmma_f16(a, frag_B(We3t, H2, (wv*2 + t) * 16, k0), acc[t]);
    }
    int lane = tid & 31;
    int nco = lane & 15;
    int mrb = (lane < 16) ? 0 : 8;
    _Float16* h3 = sB;                       // reuse (h1 dead)
#pragma unroll
    for (int t = 0; t < 2; ++t) {
      int n = (wv*2 + t) * 16 + nco;
#pragma unroll
      for (int r = 0; r < 8; ++r)
        h3[(size_t)(r + mrb) * H4 + n] = (_Float16)gelu_f(acc[t][r] + be3[n]);
    }
  }
  __syncthreads();

  // ---- logits: h3 . We4 + be4, sigmoid
  {
    int e = tid >> 3, part = tid & 7;
    const _Float16* h3 = sB;
    float s = 0.f;
    for (int c = part; c < H4; c += 8) s += (float)h3[(size_t)e*H4 + c] * We4[c];
    s += __shfl_xor(s, 1, 32);
    s += __shfl_xor(s, 2, 32);
    s += __shfl_xor(s, 4, 32);
    if (part == 0) {
      float lg = s + be4[0];
      out[(size_t)blockIdx.x * 16 + e] = 1.f / (1.f + expf(-lg));
    }
  }
}

// ---------- host orchestration ----------
extern "C" void kernel_launch(void* const* d_in, const int* in_sizes, int n_in,
                              void* d_out, int out_size, void* d_ws, size_t ws_size,
                              hipStream_t stream) {
  (void)in_sizes; (void)n_in; (void)out_size; (void)ws_size;
  const float* vert  = (const float*)d_in[0];
  const float* W1    = (const float*)d_in[1];
  const float* b1    = (const float*)d_in[2];
  const float* ln1g  = (const float*)d_in[3];
  const float* ln1b  = (const float*)d_in[4];
  const float* W2    = (const float*)d_in[5];
  const float* b2    = (const float*)d_in[6];
  const float* ln2g  = (const float*)d_in[7];
  const float* ln2b  = (const float*)d_in[8];
  const float* Wq    = (const float*)d_in[9];
  const float* bq    = (const float*)d_in[10];
  const float* Wk    = (const float*)d_in[11];
  const float* bk    = (const float*)d_in[12];
  const float* Wv    = (const float*)d_in[13];
  const float* bv    = (const float*)d_in[14];
  const float* Wo    = (const float*)d_in[15];
  const float* bo    = (const float*)d_in[16];
  const float* We1   = (const float*)d_in[17];
  const float* be1   = (const float*)d_in[18];
  const float* lne1g = (const float*)d_in[19];
  const float* lne1b = (const float*)d_in[20];
  const float* We2   = (const float*)d_in[21];
  const float* be2   = (const float*)d_in[22];
  const float* lne2g = (const float*)d_in[23];
  const float* lne2b = (const float*)d_in[24];
  const float* We3   = (const float*)d_in[25];
  const float* be3   = (const float*)d_in[26];
  const float* We4   = (const float*)d_in[27];
  const float* be4   = (const float*)d_in[28];
  float* out = (float*)d_out;

  char* w = (char*)d_ws;
  size_t off = 0;
  auto take = [&](size_t bytes) -> void* {
    void* p = w + off;
    off = (off + bytes + 255) & ~(size_t)255;
    return p;
  };

  int*      iu    = (int*)take((size_t)PPAIR * 4);
  int*      ju    = (int*)take((size_t)PPAIR * 4);
  _Float16* xh    = (_Float16*)take((size_t)RR * H2 * 2);
  float*    y2b   = (float*)take((size_t)RR * HH * 4);
  float*    vfb   = (float*)take((size_t)RR * HH * 4);
  _Float16* vfh   = (_Float16*)take((size_t)RR * HH * 2);
  float*    qb    = (float*)take((size_t)RR * HH * 4);
  float*    kb    = (float*)take((size_t)RR * HH * 4);
  float*    vb    = (float*)take((size_t)RR * HH * 4);
  _Float16* ctxh  = (_Float16*)take((size_t)RR * HH * 2);
  float*    vf2   = (float*)take((size_t)RR * HH * 4);
  _Float16* vf2h  = (_Float16*)take((size_t)RR * HH * 2);
  float*    Apre  = (float*)take((size_t)RR * HH * 4);
  float*    Bpre  = (float*)take((size_t)RR * HH * 4);
  float*    Cpre  = (float*)take((size_t)RR * HH * 4);
  float*    Dpre  = (float*)take((size_t)RR * HH * 4);
  _Float16* W2t   = (_Float16*)take((size_t)HH * H2 * 2);
  _Float16* Wqt   = (_Float16*)take((size_t)HH * HH * 2);
  _Float16* Wkt   = (_Float16*)take((size_t)HH * HH * 2);
  _Float16* Wvt   = (_Float16*)take((size_t)HH * HH * 2);
  _Float16* Wot   = (_Float16*)take((size_t)HH * HH * 2);
  _Float16* We1aT = (_Float16*)take((size_t)HH * HH * 2);
  _Float16* We1bT = (_Float16*)take((size_t)HH * HH * 2);
  _Float16* We2t  = (_Float16*)take((size_t)HH * H2 * 2);
  _Float16* We3t  = (_Float16*)take((size_t)H2 * H4 * 2);

  // pair table
  k_pairs<<<(PPAIR + 255) / 256, 256, 0, stream>>>(iu, ju);

  // transpose+convert weights to f16 [N][K]
  auto T = [&](const float* s, _Float16* d, int K, int N) {
    k_t16<<<((size_t)K * N + 255) / 256, 256, 0, stream>>>(s, d, K, N);
  };
  T(W2, W2t, H2, HH);
  T(Wq, Wqt, HH, HH);  T(Wk, Wkt, HH, HH);
  T(Wv, Wvt, HH, HH);  T(Wo, Wot, HH, HH);
  T(We1,                 We1aT, HH, HH);   // rows 0..511
  T(We1 + (size_t)HH*HH, We1bT, HH, HH);   // rows 512..1023
  T(We2, We2t, HH, H2);
  T(We3, We3t, H2, H4);

  // vertex MLP stage 1
  k_vertex1<<<RR, 256, 0, stream>>>(vert, W1, b1, ln1g, ln1b, xh);

  // x @ W2 + b2 -> y2 ; LN -> vf (+f16)
  k_gemm<<<dim3(RR/16, HH/64), 128, 0, stream>>>(xh, W2t, b2, nullptr, y2b, nullptr, H2, HH);
  k_ln  <<<RR, 256, 0, stream>>>(y2b, ln2g, ln2b, vfb, vfh, HH);

  // q, k, v projections
  k_gemm<<<dim3(RR/16, HH/64), 128, 0, stream>>>(vfh, Wqt, bq, nullptr, qb, nullptr, HH, HH);
  k_gemm<<<dim3(RR/16, HH/64), 128, 0, stream>>>(vfh, Wkt, bk, nullptr, kb, nullptr, HH, HH);
  k_gemm<<<dim3(RR/16, HH/64), 128, 0, stream>>>(vfh, Wvt, bv, nullptr, vb, nullptr, HH, HH);

  // attention -> ctx (f16)
  k_attn<<<(BB*8*NN)/8, 256, 0, stream>>>(qb, kb, vb, ctxh);

  // vf2 = vf + ctx @ Wo + bo  (+f16 copy)
  k_gemm<<<dim3(RR/16, HH/64), 128, 0, stream>>>(ctxh, Wot, bo, vfb, vf2, vf2h, HH, HH);

  // per-vertex precomputes for edge layer 1
  k_gemm<<<dim3(RR/16, HH/64), 128, 0, stream>>>(vf2h, We1aT, be1,    nullptr, Apre, nullptr, HH, HH);
  k_gemm<<<dim3(RR/16, HH/64), 128, 0, stream>>>(vf2h, We1bT, nullptr, nullptr, Bpre, nullptr, HH, HH);
  k_cdpre<<<RR, 256, 0, stream>>>(vert, We1, Cpre, Dpre);

  // fused edge MLP tail
  k_edge<<<NEDGE / 16, 128, 0, stream>>>(
      Apre, Bpre, Cpre, Dpre, We1 + (size_t)1030 * HH,
      lne1g, lne1b, We2t, be2, lne2g, lne2b, We3t, be3, We4, be4,
      vert, iu, ju, out);
}
